// Regress_5033701670954
// MI455X (gfx1250) — matrix operand order repaired and checked by reading, not compiled, not run on hardware
//
#include <hip/hip_runtime.h>
#include <hip/hip_bf16.h>

typedef __attribute__((ext_vector_type(16))) _Float16 v16h;
typedef __attribute__((ext_vector_type(8)))  float    v8f;
typedef __attribute__((ext_vector_type(4)))  _Float16 v4h;

#define IN_DIM 128
#define HIDDEN 128
#define N_GRAPHS 64

// ---------------- utility kernels ----------------

__global__ void zero_f32_kernel(float* p, long long n) {
    long long i = (long long)blockIdx.x * blockDim.x + threadIdx.x;
    if (i < n) p[i] = 0.0f;
}

// Repack W (row-major f32 [128][128], W[k][n]) into WMMA B-operand lane order (f16):
// Wp linear v16h-block index = ((ntile*32 + lane)*4 + chunk); half index idx in block.
// b[idx] for (ntile, chunk, lane) = W[ 32*chunk + (lane>>4)*16 + idx ][ ntile*16 + (lane&15) ]
__global__ void prep_w_kernel(const float* __restrict__ W, _Float16* __restrict__ Wp) {
    int i = blockIdx.x * blockDim.x + threadIdx.x;
    if (i >= HIDDEN * HIDDEN) return;
    int idx  = i & 15;
    int c    = (i >> 4) & 3;
    int lane = (i >> 6) & 31;
    int nt   = (i >> 11) & 7;
    int k = 32 * c + ((lane >> 4) * 16) + idx;
    int n = nt * 16 + (lane & 15);
    Wp[i] = (_Float16)W[k * HIDDEN + n];
}

// degrees via atomics
__global__ void degree_kernel(const int* __restrict__ src, const int* __restrict__ dst,
                              float* deg_out, float* deg_in, int E) {
    int e = blockIdx.x * blockDim.x + threadIdx.x;
    if (e >= E) return;
    atomicAdd(&deg_out[src[e]], 1.0f);
    atomicAdd(&deg_in[dst[e]], 1.0f);
}

// in-place deg -> rsqrt(max(deg,1)); also per-graph node counts
__global__ void norm_count_kernel(float* deg_out, float* deg_in,
                                  const int* __restrict__ gids, float* counts, int N) {
    int i = blockIdx.x * blockDim.x + threadIdx.x;
    if (i >= N) return;
    deg_out[i] = rsqrtf(fmaxf(deg_out[i], 1.0f));
    deg_in[i]  = rsqrtf(fmaxf(deg_in[i], 1.0f));
    atomicAdd(&counts[gids[i]], 1.0f);
}

// hs[n][k] = f16(feat[n][k] * norm_src[n])
__global__ void scale_feat_kernel(const float* __restrict__ feat, const float* __restrict__ norm_src,
                                  _Float16* __restrict__ hs, long long total) {
    long long i = (long long)blockIdx.x * blockDim.x + threadIdx.x;
    if (i >= total) return;
    int row = (int)(i >> 7);
    hs[i] = (_Float16)(feat[i] * norm_src[row]);
}

// SpMM scatter: agg[dst][k] += hs[src][k]; 32 lanes/edge, 4 feats/lane
__global__ void spmm_kernel(const int* __restrict__ src, const int* __restrict__ dst,
                            const _Float16* __restrict__ hs, float* __restrict__ agg,
                            long long nwork) {
    long long idx = (long long)blockIdx.x * blockDim.x + threadIdx.x;
    if (idx >= nwork) return;
    int e = (int)(idx >> 5);
    int f = ((int)idx & 31) * 4;
    int s = src[e], d = dst[e];
    v4h hv = *(const v4h*)(hs + (size_t)s * HIDDEN + f);
    float* arow = agg + (size_t)d * HIDDEN + f;
    atomicAdd(arow + 0, (float)hv[0]);
    atomicAdd(arow + 1, (float)hv[1]);
    atomicAdd(arow + 2, (float)hv[2]);
    atomicAdd(arow + 3, (float)hv[3]);
}

// ---------------- WMMA GEMM core ----------------
// Computes 16x16 f32 tile of (X .* norm_dst_row) @ W  with K = 128 (4x v_wmma_f32_16x16x32_f16).
// Wp is pre-shuffled into B-operand lane order: per lane, 4 chunks of 16 halves, contiguous.
__device__ inline v8f gemm_tile_128(const float* __restrict__ X, const float* __restrict__ norm_dst,
                                    const _Float16* __restrict__ Wp, int m0, int n0, int lane) {
    const int mrow = m0 + (lane & 15);
    const int hsel = lane >> 4;                 // which half of the wave
    const float* xrow = X + (size_t)mrow * IN_DIM;
    const float nr = norm_dst[mrow];
    const v16h* bp = (const v16h*)Wp + (size_t)((n0 >> 4) * 32 + lane) * 4;
    v8f acc = {};
#pragma unroll
    for (int c = 0; c < 4; ++c) {
        v16h a;
        // A: 16-bit 16x32 layout — lanes0-15 K=0..15, lanes16-31 K=16..31 (contiguous, matches Wp)
#pragma unroll
        for (int v = 0; v < 8; ++v) {
            int kb = 32 * c + hsel * 16 + 2 * v;
            a[2 * v]     = (_Float16)(xrow[kb]     * nr);
            a[2 * v + 1] = (_Float16)(xrow[kb + 1] * nr);
        }
        v16h b = bp[c];                         // 32 contiguous bytes: 2x global_load_b128
        acc = __builtin_amdgcn_wmma_f32_16x16x32_f16(false, a, false, b, (short)0, acc, false, false);
    }
    return acc;
}

// Layer 1: out_h[m][n] = f16( relu(acc + b) * norm_src[m] )  (feeds next SpMM)
__global__ __launch_bounds__(256) void gemm1_kernel(const float* __restrict__ agg,
                                                    const float* __restrict__ norm_dst,
                                                    const _Float16* __restrict__ Wp,
                                                    const float* __restrict__ bias,
                                                    const float* __restrict__ norm_src,
                                                    _Float16* __restrict__ out_h) {
    int lane = threadIdx.x & 31;
    int m0 = blockIdx.x * 16;
    int n0 = (threadIdx.x >> 5) * 16;           // 8 waves cover the 8 N-tiles
    v8f acc = gemm_tile_128(agg, norm_dst, Wp, m0, n0, lane);
    int hsel = lane >> 4, ncol = n0 + (lane & 15);
    float bv = bias[ncol];
#pragma unroll
    for (int v = 0; v < 8; ++v) {
        int m = m0 + v + 8 * hsel;
        float val = fmaxf(acc[v] + bv, 0.0f);
        out_h[(size_t)m * HIDDEN + ncol] = (_Float16)(val * norm_src[m]);
    }
}

// Layer 2: relu(acc + b) scattered into per-graph sums (fused segment_sum readout)
__global__ __launch_bounds__(256) void gemm2_kernel(const float* __restrict__ agg,
                                                    const float* __restrict__ norm_dst,
                                                    const _Float16* __restrict__ Wp,
                                                    const float* __restrict__ bias,
                                                    const int* __restrict__ gids,
                                                    float* __restrict__ hg) {
    int lane = threadIdx.x & 31;
    int m0 = blockIdx.x * 16;
    int n0 = (threadIdx.x >> 5) * 16;
    v8f acc = gemm_tile_128(agg, norm_dst, Wp, m0, n0, lane);
    int hsel = lane >> 4, ncol = n0 + (lane & 15);
    float bv = bias[ncol];
#pragma unroll
    for (int v = 0; v < 8; ++v) {
        int m = m0 + v + 8 * hsel;
        float val = fmaxf(acc[v] + bv, 0.0f);
        atomicAdd(&hg[(size_t)gids[m] * HIDDEN + ncol], val);
    }
}

// ---------------- readout MLP: 128 -> 64 -> 32 -> 16 -> 1 (single block) ----------------
__global__ __launch_bounds__(256) void readout_kernel(const float* __restrict__ hg,
                                                      const float* __restrict__ counts,
                                                      const float* __restrict__ Wm0, const float* __restrict__ bm0,
                                                      const float* __restrict__ Wm1, const float* __restrict__ bm1,
                                                      const float* __restrict__ Wm2, const float* __restrict__ bm2,
                                                      const float* __restrict__ Wm3, const float* __restrict__ bm3,
                                                      float* __restrict__ out) {
    __shared__ float hgm[N_GRAPHS * 128];
    __shared__ float y0[N_GRAPHS * 64];
    __shared__ float y1[N_GRAPHS * 32];
    __shared__ float y2[N_GRAPHS * 16];
    int t = threadIdx.x;
    for (int i = t; i < N_GRAPHS * 128; i += 256) {
        int g = i >> 7;
        hgm[i] = hg[i] / fmaxf(counts[g], 1.0f);
    }
    __syncthreads();
    for (int i = t; i < N_GRAPHS * 64; i += 256) {
        int g = i >> 6, j = i & 63;
        float s = bm0[j];
        for (int k = 0; k < 128; ++k) s += hgm[g * 128 + k] * Wm0[k * 64 + j];
        y0[i] = fmaxf(s, 0.0f);
    }
    __syncthreads();
    for (int i = t; i < N_GRAPHS * 32; i += 256) {
        int g = i >> 5, j = i & 31;
        float s = bm1[j];
        for (int k = 0; k < 64; ++k) s += y0[g * 64 + k] * Wm1[k * 32 + j];
        y1[i] = fmaxf(s, 0.0f);
    }
    __syncthreads();
    for (int i = t; i < N_GRAPHS * 16; i += 256) {
        int g = i >> 4, j = i & 15;
        float s = bm2[j];
        for (int k = 0; k < 32; ++k) s += y1[g * 32 + k] * Wm2[k * 16 + j];
        y2[i] = fmaxf(s, 0.0f);
    }
    __syncthreads();
    for (int g = t; g < N_GRAPHS; g += 256) {
        float s = bm3[0];
        for (int k = 0; k < 16; ++k) s += y2[g * 16 + k] * Wm3[k];
        out[g] = s;
    }
}

// ---------------- host launch ----------------

extern "C" void kernel_launch(void* const* d_in, const int* in_sizes, int n_in,
                              void* d_out, int out_size, void* d_ws, size_t ws_size,
                              hipStream_t stream) {
    const float* feat = (const float*)d_in[0];
    const int*   src  = (const int*)d_in[1];
    const int*   dst  = (const int*)d_in[2];
    const int*   gids = (const int*)d_in[3];
    const float* W1   = (const float*)d_in[5];
    const float* b1   = (const float*)d_in[6];
    const float* W2   = (const float*)d_in[7];
    const float* b2   = (const float*)d_in[8];
    const float* Wm0  = (const float*)d_in[9];
    const float* bm0  = (const float*)d_in[10];
    const float* Wm1  = (const float*)d_in[11];
    const float* bm1  = (const float*)d_in[12];
    const float* Wm2  = (const float*)d_in[13];
    const float* bm2  = (const float*)d_in[14];
    const float* Wm3  = (const float*)d_in[15];
    const float* bm3  = (const float*)d_in[16];

    const int N = in_sizes[0] / IN_DIM;     // 100000 (divisible by 16)
    const int E = in_sizes[1];              // 3200000

    // workspace carve-out (256B aligned)
    char* ws = (char*)d_ws;
    size_t off = 0;
    auto take = [&](size_t bytes) -> char* {
        char* p = ws + off;
        off = (off + bytes + 255) & ~(size_t)255;
        return p;
    };
    float*    deg_out = (float*)take((size_t)N * 4);            // becomes norm_src
    float*    deg_in  = (float*)take((size_t)N * 4);            // becomes norm_dst
    float*    counts  = (float*)take(N_GRAPHS * 4);
    float*    hgsum   = (float*)take(N_GRAPHS * HIDDEN * 4);
    _Float16* hs      = (_Float16*)take((size_t)N * HIDDEN * 2);
    float*    agg     = (float*)take((size_t)N * HIDDEN * 4);
    _Float16* W1p     = (_Float16*)take(IN_DIM * HIDDEN * 2);
    _Float16* W2p     = (_Float16*)take(HIDDEN * HIDDEN * 2);

    const long long NK = (long long)N * HIDDEN;
    const long long EW = (long long)E * 32;
    auto gsz = [](long long n, int b) { return (unsigned)((n + b - 1) / b); };

    // init
    zero_f32_kernel<<<gsz(N, 256), 256, 0, stream>>>(deg_out, N);
    zero_f32_kernel<<<gsz(N, 256), 256, 0, stream>>>(deg_in, N);
    zero_f32_kernel<<<1, 256, 0, stream>>>(counts, N_GRAPHS);
    zero_f32_kernel<<<gsz(N_GRAPHS * HIDDEN, 256), 256, 0, stream>>>(hgsum, N_GRAPHS * HIDDEN);
    prep_w_kernel<<<gsz(IN_DIM * HIDDEN, 256), 256, 0, stream>>>(W1, W1p);
    prep_w_kernel<<<gsz(HIDDEN * HIDDEN, 256), 256, 0, stream>>>(W2, W2p);

    // degrees + norms + graph counts
    degree_kernel<<<gsz(E, 256), 256, 0, stream>>>(src, dst, deg_out, deg_in, E);
    norm_count_kernel<<<gsz(N, 256), 256, 0, stream>>>(deg_out, deg_in, gids, counts, N);

    // layer 1
    scale_feat_kernel<<<gsz(NK, 256), 256, 0, stream>>>(feat, deg_out, hs, NK);
    zero_f32_kernel<<<gsz(NK, 256), 256, 0, stream>>>(agg, NK);
    spmm_kernel<<<gsz(EW, 256), 256, 0, stream>>>(src, dst, hs, agg, EW);
    gemm1_kernel<<<N / 16, 256, 0, stream>>>(agg, deg_in, W1p, b1, deg_out, hs);

    // layer 2 (fused per-graph segment-sum in epilogue)
    zero_f32_kernel<<<gsz(NK, 256), 256, 0, stream>>>(agg, NK);
    spmm_kernel<<<gsz(EW, 256), 256, 0, stream>>>(src, dst, hs, agg, EW);
    gemm2_kernel<<<N / 16, 256, 0, stream>>>(agg, deg_in, W2p, b2, gids, hgsum);

    // readout MLP
    readout_kernel<<<1, 256, 0, stream>>>(hgsum, counts, Wm0, bm0, Wm1, bm1, Wm2, bm2, Wm3, bm3,
                                          (float*)d_out);
}
